// RadialReceptiveField_19808389169825
// MI455X (gfx1250) — matrix operand order repaired and checked by reading, not compiled, not run on hardware
//
#include <hip/hip_runtime.h>

// Problem constants (match reference)
#define BB_   4
#define CCH   64
#define IMG_H 128
#define IMG_W 128
#define HW    16384      // IMG_H*IMG_W
#define PP    8
#define RR    5
#define K1    512        // C*P
#define TM    64         // pixels per block
#define NTHREADS 256

typedef __bf16 bf16;
typedef __attribute__((ext_vector_type(8)))  __bf16 v8bf;
typedef __attribute__((ext_vector_type(16))) __bf16 v16bf;
typedef __attribute__((ext_vector_type(8)))  float  v8f;
typedef __attribute__((ext_vector_type(4)))  unsigned int v4u;
typedef __attribute__((ext_vector_type(8)))  int v8i;
typedef __attribute__((ext_vector_type(4)))  int v4i;

union V16 { v16bf v; v8bf h[2]; };

__device__ __forceinline__ bf16 f2bf(float f) {
  // round-to-nearest-even f32 -> bf16
  unsigned u = __builtin_bit_cast(unsigned, f);
  unsigned r = (u + 0x7FFFu + ((u >> 16) & 1u)) >> 16;
  unsigned short s = (unsigned short)r;
  return __builtin_bit_cast(bf16, s);
}

// ---- TDM availability ----
#if defined(__has_builtin)
#if __has_builtin(__builtin_amdgcn_tensor_load_to_lds) && __has_builtin(__builtin_amdgcn_s_wait_tensorcnt)
#define HAVE_TDM 1
#endif
#endif

#ifdef HAVE_TDM
// 1-D contiguous DMA: global -> LDS, ndwords 4-byte elements (ndwords <= 65535).
// D# group0: count=1 | lds_addr | global_addr[56:0] | type=2
// D# group1: data_size=4B; tensor_dim0=tile_dim0=ndwords; dim1=1; stride0=ndwords
__device__ __forceinline__ void tdm_load_1d(unsigned lds_off, const void* gsrc, unsigned ndwords) {
  unsigned long long ga = (unsigned long long)gsrc;
  v4u g0;
  g0.x = 1u;                                                    // count=1, is_restore=0, gather=0
  g0.y = lds_off;                                               // lds_addr (bytes)
  g0.z = (unsigned)ga;                                          // global_addr[31:0]
  g0.w = (unsigned)((ga >> 32) & 0x01FFFFFFull) | 0x80000000u;  // global_addr[56:32] | type=2
  v8i g1;
  g1[0] = (int)(2u << 16);                                      // wg_mask=0, data_size=2 (4B)
  g1[1] = (int)((ndwords & 0xFFFFu) << 16);                     // abar_addr=0 | tensor_dim0[15:0]
  g1[2] = (int)((ndwords >> 16) | (1u << 16));                  // tensor_dim0[31:16] | tensor_dim1=1
  g1[3] = (int)((ndwords & 0xFFFFu) << 16);                     // tensor_dim1 hi=0 | tile_dim0
  g1[4] = 1;                                                    // tile_dim1=1, tile_dim2=0
  g1[5] = (int)ndwords;                                         // tensor_dim0_stride[31:0]
  g1[6] = 0;                                                    // stride0 hi | tensor_dim1_stride lo
  g1[7] = 0;
  v4i z4 = {0, 0, 0, 0};
#if __clang_major__ >= 23
  v8i z8 = {0, 0, 0, 0, 0, 0, 0, 0};
  __builtin_amdgcn_tensor_load_to_lds(g0, g1, z4, z4, z8, 0);
#else
  __builtin_amdgcn_tensor_load_to_lds(g0, g1, z4, z4, 0);
#endif
}
#endif

// ---- LDS layout (bytes) ----
#define OFF_FEAT   0            // TM x 512 bf16          = 65536
#define OFF_W1     65536        // 32768 bf16 (B-layout)  = 65536
#define OFF_W2     131072       // 4096 bf16  (B-layout)  = 8192
#define OFF_HBF    139264       // 64x64 bf16             = 8192
#define OFF_HF     147456       // 64x65 f32 (padded)     = 16640
#define OFF_PAR    164096       // 4*64 f32               = 1024
#define SMEM_BYTES 165120

// ================= Prep kernel: shuffle f32 weights into WMMA-B bf16 lane layout =================
// layout: w[((ks*4+nt)*32 + lane)*16 + e] = W[(ks*32 + (lane>>4)*16 + e)*64 + nt*16 + (lane&15)]
__global__ __launch_bounds__(NTHREADS)
void prep_weights(const float* __restrict__ W1, const float* __restrict__ W2,
                  bf16* __restrict__ w1p, bf16* __restrict__ w2p)
{
  int idx = blockIdx.x * NTHREADS + threadIdx.x;
  if (idx < RR * K1 * 64) {            // 5 x 32768
    int r = idx >> 15;
    int t = idx & 32767;
    int e  = t & 15;
    int ln = (t >> 4) & 31;
    int nt = (t >> 9) & 3;
    int ks = t >> 11;
    int k  = ks * 32 + (ln >> 4) * 16 + e;
    int n  = nt * 16 + (ln & 15);
    w1p[idx] = f2bf(W1[(size_t)r * K1 * 64 + k * 64 + n]);
  }
  if (idx < RR * 64 * 64) {            // 5 x 4096
    int r = idx >> 12;
    int t = idx & 4095;
    int e  = t & 15;
    int ln = (t >> 4) & 31;
    int nt = (t >> 9) & 3;
    int ks = t >> 11;
    int k  = ks * 32 + (ln >> 4) * 16 + e;
    int n  = nt * 16 + (ln & 15);
    w2p[idx] = f2bf(W2[(size_t)r * 64 * 64 + k * 64 + n]);
  }
}

// ================= Main kernel =================
__global__ __launch_bounds__(NTHREADS)
void rrf_kernel(const float* __restrict__ x,
                const bf16* __restrict__ w1p,
                const float* __restrict__ b1,
                const float* __restrict__ g1,
                const float* __restrict__ bt1,
                const bf16* __restrict__ w2p,
                const float* __restrict__ b2,
                float* __restrict__ out)
{
  extern __shared__ char smem[];
  bf16*  featS = (bf16*)(smem + OFF_FEAT);
  bf16*  w1S   = (bf16*)(smem + OFF_W1);
  bf16*  w2S   = (bf16*)(smem + OFF_W2);
  bf16*  hbfS  = (bf16*)(smem + OFF_HBF);
  float* hF    = (float*)(smem + OFF_HF);   // stride 65: bank-conflict-free column reads
  float* b1S   = (float*)(smem + OFF_PAR);
  float* g1S   = b1S + 64;
  float* bt1S  = b1S + 128;
  float* b2S   = b1S + 192;

  const int tid   = threadIdx.x;
  const int lane  = tid & 31;
  const int wv    = tid >> 5;
  const int mtile = wv & 3;        // 4 M-tiles of 16 rows
  const int npair = wv >> 2;       // waves 0-3 -> N-tiles {0,1}; waves 4-7 -> {2,3}
  const int half  = lane >> 4;
  const int l16   = lane & 15;
  const int nt0   = npair * 2 + 0;
  const int nt1   = npair * 2 + 1;
  const int n0    = nt0 * 16 + l16;
  const int n1    = nt1 * 16 + l16;
  const int arow  = mtile * 16 + l16;   // A-matrix row this lane supplies

  const int pix0 = blockIdx.x * TM;     // 64 consecutive pixels, same batch, same image row
  const int bb   = pix0 >> 14;          // / HW
  const int rem0 = pix0 & (HW - 1);
  const int hrow = rem0 >> 7;           // / IMG_W
  const int w0   = rem0 & (IMG_W - 1);
  const float* xb = x + (size_t)bb * CCH * HW;

#ifdef HAVE_TDM
  const unsigned ldsOffW1 = (unsigned)(unsigned long long)(void*)w1S;
  const unsigned ldsOffW2 = (unsigned)(unsigned long long)(void*)w2S;
#endif

  v8f acc0 = {0.f,0.f,0.f,0.f,0.f,0.f,0.f,0.f};
  v8f acc1 = {0.f,0.f,0.f,0.f,0.f,0.f,0.f,0.f};

  for (int r = 0; r < RR; ++r) {
    // ---- Stage weights: TDM DMA (overlaps with the sampling gather below) ----
#ifdef HAVE_TDM
    if (wv == 0) {
      tdm_load_1d(ldsOffW1, w1p + (size_t)r * K1 * 64, (K1 * 64 * 2) / 4);   // 64KB -> 16384 dw
      tdm_load_1d(ldsOffW2, w2p + (size_t)r * 64 * 64, (64 * 64 * 2) / 4);   // 8KB  ->  2048 dw
    }
#endif
    if (tid < 64) {
      b1S[tid]  = b1 [r * 64 + tid];
      g1S[tid]  = g1 [r * 64 + tid];
      bt1S[tid] = bt1[r * 64 + tid];
      b2S[tid]  = b2 [r * 64 + tid];
    }

    // ---- Ring sampling -> featS[row*512 + c*8 + p] (bf16) ----
    {
      const float rf = (float)r;
      for (int t = tid; t < TM * PP; t += NTHREADS) {
        int row = t >> 3;
        int p   = t & 7;
        float ang = (float)p * 0.78539816339744831f;  // 2*pi*p/8
        float di = rf * __sinf(ang);
        float dj = rf * __cosf(ang);
        float fy = floorf(di), fx = floorf(dj);
        float wy = di - fy,    wx = dj - fx;
        int y0 = hrow + (int)fy;
        int y1 = y0 + 1;
        int xw = w0 + row;
        int x0i = xw + (int)fx;
        int x1i = x0i + 1;
        bool vy0 = (y0 >= 0) && (y0 < IMG_H);
        bool vy1 = (y1 >= 0) && (y1 < IMG_H);
        bool vx0 = (x0i >= 0) && (x0i < IMG_W);
        bool vx1 = (x1i >= 0) && (x1i < IMG_W);
        float w00 = (vy0 && vx0) ? (1.f - wy) * (1.f - wx) : 0.f;
        float w01 = (vy0 && vx1) ? (1.f - wy) * wx         : 0.f;
        float w10 = (vy1 && vx0) ? wy * (1.f - wx)         : 0.f;
        float w11 = (vy1 && vx1) ? wy * wx                 : 0.f;
        int yc0 = min(max(y0, 0), IMG_H - 1), yc1 = min(max(y1, 0), IMG_H - 1);
        int xc0 = min(max(x0i, 0), IMG_W - 1), xc1 = min(max(x1i, 0), IMG_W - 1);
        const float* p00 = xb + yc0 * IMG_W + xc0;
        const float* p01 = xb + yc0 * IMG_W + xc1;
        const float* p10 = xb + yc1 * IMG_W + xc0;
        const float* p11 = xb + yc1 * IMG_W + xc1;
        bf16* dst = featS + row * K1 + p;
        #pragma unroll 4
        for (int c = 0; c < CCH; ++c) {
          int off = c * HW;
          float v = w00 * p00[off] + w01 * p01[off] + w10 * p10[off] + w11 * p11[off];
          dst[c * PP] = f2bf(v);
        }
      }
    }

#ifdef HAVE_TDM
    if (wv == 0) __builtin_amdgcn_s_wait_tensorcnt(0);   // TENSORcnt==0: DMA done
#else
    // Fallback: vectorized copy of pre-shuffled bf16 weights into LDS
    for (int i = tid; i < (K1 * 64 * 2) / 16; i += NTHREADS)
      ((uint4*)w1S)[i] = ((const uint4*)(w1p + (size_t)r * K1 * 64))[i];
    for (int i = tid; i < (64 * 64 * 2) / 16; i += NTHREADS)
      ((uint4*)w2S)[i] = ((const uint4*)(w2p + (size_t)r * 64 * 64))[i];
#endif
    __syncthreads();

    // ---- GEMM1: [64x512] x [512x64] via v_wmma_f32_16x16x32_bf16 ----
    v8f c0 = {0.f,0.f,0.f,0.f,0.f,0.f,0.f,0.f};
    v8f c1 = {0.f,0.f,0.f,0.f,0.f,0.f,0.f,0.f};
    #pragma unroll
    for (int ks = 0; ks < 16; ++ks) {
      int kb = ks * 32 + half * 16;  // contiguous K order (B packed to match): lane half picks K 0-15 vs 16-31
      V16 a;
      a.h[0] = *(const v8bf*)(featS + arow * K1 + kb);
      a.h[1] = *(const v8bf*)(featS + arow * K1 + kb + 8);
      V16 bA, bB;
      const v8bf* bp0 = (const v8bf*)(w1S + ((ks * 4 + nt0) * 32 + lane) * 16);
      const v8bf* bp1 = (const v8bf*)(w1S + ((ks * 4 + nt1) * 32 + lane) * 16);
      bA.h[0] = bp0[0]; bA.h[1] = bp0[1];
      bB.h[0] = bp1[0]; bB.h[1] = bp1[1];
      c0 = __builtin_amdgcn_wmma_f32_16x16x32_bf16(false, a.v, false, bA.v, (short)0, c0, false, false);
      c1 = __builtin_amdgcn_wmma_f32_16x16x32_bf16(false, a.v, false, bB.v, (short)0, c1, false, false);
    }
    // scatter h (+bias); C layout: vgpr v -> row mtile*16 + half*8 + v, col n
    {
      float bv0 = b1S[n0], bv1 = b1S[n1];
      #pragma unroll
      for (int v = 0; v < 8; ++v) {
        int rrow = mtile * 16 + half * 8 + v;
        hF[rrow * 65 + n0] = c0[v] + bv0;
        hF[rrow * 65 + n1] = c1[v] + bv1;
      }
    }
    __syncthreads();

    // ---- LayerNorm(64) + exact GELU, requantize to bf16 ----
    if (tid < TM) {
      const float* hr = hF + tid * 65;
      float mu = 0.f;
      #pragma unroll 8
      for (int c = 0; c < 64; ++c) mu += hr[c];
      mu *= (1.f / 64.f);
      float var = 0.f;
      #pragma unroll 8
      for (int c = 0; c < 64; ++c) { float d = hr[c] - mu; var += d * d; }
      var *= (1.f / 64.f);
      float rstd = rsqrtf(var + 1e-5f);
      bf16* hb = hbfS + tid * 64;
      #pragma unroll 4
      for (int c = 0; c < 64; ++c) {
        float v = (hr[c] - mu) * rstd * g1S[c] + bt1S[c];
        float ge = 0.5f * v * (1.f + erff(v * 0.70710678118654752f));
        hb[c] = f2bf(ge);
      }
    }
    __syncthreads();

    // ---- GEMM2: [64x64] x [64x64], accumulate across radii ----
    #pragma unroll
    for (int ks = 0; ks < 2; ++ks) {
      int kb = ks * 32 + half * 16;
      V16 a;
      a.h[0] = *(const v8bf*)(hbfS + arow * 64 + kb);
      a.h[1] = *(const v8bf*)(hbfS + arow * 64 + kb + 8);
      V16 bA, bB;
      const v8bf* bp0 = (const v8bf*)(w2S + ((ks * 4 + nt0) * 32 + lane) * 16);
      const v8bf* bp1 = (const v8bf*)(w2S + ((ks * 4 + nt1) * 32 + lane) * 16);
      bA.h[0] = bp0[0]; bA.h[1] = bp0[1];
      bB.h[0] = bp1[0]; bB.h[1] = bp1[1];
      acc0 = __builtin_amdgcn_wmma_f32_16x16x32_bf16(false, a.v, false, bA.v, (short)0, acc0, false, false);
      acc1 = __builtin_amdgcn_wmma_f32_16x16x32_bf16(false, a.v, false, bB.v, (short)0, acc1, false, false);
    }
    {
      float bv0 = b2S[n0], bv1 = b2S[n1];
      #pragma unroll
      for (int v = 0; v < 8; ++v) { acc0[v] += bv0; acc1[v] += bv1; }
    }
    __syncthreads();   // protect feat/w1/w2/hbf before next radius rewrites them
  }

  // ---- Epilogue: sigmoid gate, coalesced store ----
  #pragma unroll
  for (int v = 0; v < 8; ++v) {
    int rrow = mtile * 16 + half * 8 + v;
    hF[rrow * 65 + n0] = acc0[v];
    hF[rrow * 65 + n1] = acc1[v];
  }
  __syncthreads();
  for (int idx = tid; idx < TM * CCH; idx += NTHREADS) {
    int pix = idx & 63;        // fastest across lanes -> coalesced global access
    int c   = idx >> 6;
    size_t g = (size_t)(bb * CCH + c) * HW + rem0 + pix;
    float a = hF[pix * 65 + c];           // stride 65 -> conflict-free
    float s = 1.f / (1.f + __expf(-a));
    out[g] = x[g] * s;
  }
}

extern "C" void kernel_launch(void* const* d_in, const int* in_sizes, int n_in,
                              void* d_out, int out_size, void* d_ws, size_t ws_size,
                              hipStream_t stream) {
  (void)in_sizes; (void)n_in; (void)out_size; (void)ws_size;
  const float* x   = (const float*)d_in[0];
  const float* W1  = (const float*)d_in[1];
  const float* b1  = (const float*)d_in[2];
  const float* g1  = (const float*)d_in[3];
  const float* bt1 = (const float*)d_in[4];
  const float* W2  = (const float*)d_in[5];
  const float* b2  = (const float*)d_in[6];
  float* out = (float*)d_out;

  // Scratch layout: pre-shuffled bf16 weights in WMMA B-operand lane order
  bf16* w1p = (bf16*)d_ws;                                  // 5*32768 bf16 = 327680 B
  bf16* w2p = (bf16*)((char*)d_ws + RR * K1 * 64 * 2);      // 5*4096  bf16 =  40960 B

  {
    const int total = RR * K1 * 64;                         // 163840 (covers W2's 20480 too)
    const int blocks = (total + NTHREADS - 1) / NTHREADS;   // 640
    prep_weights<<<blocks, NTHREADS, 0, stream>>>(W1, W2, w1p, w2p);
  }
  {
    const int total_pixels = BB_ * HW;                      // 65536
    const int blocks = total_pixels / TM;                   // 1024
    rrf_kernel<<<blocks, NTHREADS, SMEM_BYTES, stream>>>(x, w1p, b1, g1, bt1, w2p, b2, out);
  }
}
